// ScanLSTM_730144440530
// MI455X (gfx1250) — matrix operand order repaired and checked by reading, not compiled, not run on hardware
//
#include <hip/hip_runtime.h>
#include <hip/hip_bf16.h>

// ---------------------------------------------------------------------------
// ScanLSTM for MI455X (gfx1250): bf16 WMMA recurrent steps, graph-friendly.
//   T=512, B=128, D=1024, H=1024
//   gates = [x_t, h_masked] @ [Wx; Wh] + b   (K = 2048, N = 4096, M = 128)
// ---------------------------------------------------------------------------

#define TT 512
#define BB 128
#define DD 1024
#define HH 1024
#define KTOT 2048   // D + H
#define NG   4096   // 4*H

typedef __attribute__((ext_vector_type(16))) __bf16 v16bf;
typedef __attribute__((ext_vector_type(8)))  float  v8f;

// round-to-nearest-even f32 -> bf16 (integer path, no __bf16 conversions)
static __device__ __forceinline__ unsigned short f2bf(float f) {
    union { float f; unsigned u; } v; v.f = f;
    unsigned r = v.u + 0x7FFFu + ((v.u >> 16) & 1u);
    return (unsigned short)(r >> 16);
}

static __device__ __forceinline__ v16bf make_frag(uint4 lo, uint4 hi) {
    struct S { uint4 a, b; } s{lo, hi};
    return __builtin_bit_cast(v16bf, s);
}

static __device__ __forceinline__ float sigmoidf_(float x) {
    return 1.0f / (1.0f + __expf(-x));
}
static __device__ __forceinline__ float tanhf_(float x) {
    x = fminf(fmaxf(x, -15.0f), 15.0f);
    float e = __expf(2.0f * x);
    return (e - 1.0f) / (e + 1.0f);
}

// ---------------------------------------------------------------------------
// Prep: build transposed bf16 combined weights  WcT[n][k] (n<4096, k<2048)
//   k < 1024 -> Wx[k][n] ; k >= 1024 -> Wh[k-1024][n]
// ---------------------------------------------------------------------------
__global__ __launch_bounds__(256)
void lstm_prep_weights(const float* __restrict__ Wx,
                       const float* __restrict__ Wh,
                       unsigned short* __restrict__ WcT) {
    size_t idx = (size_t)blockIdx.x * blockDim.x + threadIdx.x; // 0 .. 4096*2048-1
    int n = (int)(idx >> 11);      // / 2048
    int k = (int)(idx & 2047);
    float v = (k < DD) ? Wx[(size_t)k * NG + n]
                       : Wh[(size_t)(k - DD) * NG + n];
    WcT[idx] = f2bf(v);
}

// ---------------------------------------------------------------------------
// One recurrent step.
//   grid  = (4, 16)  : blockIdx.x = M-tile (32 rows), blockIdx.y = H-tile (64 cols)
//   block = 256 (8 waves): wave = (wr in 0..1) * 4 + (wc in 0..3)
//   wave computes 16 rows x 16 cols of each of i,f,g,o  (4 accumulators)
// ---------------------------------------------------------------------------
__global__ __launch_bounds__(256)
void lstm_step(const float* __restrict__ obs_t,          // B x D
               const unsigned char* __restrict__ resets_t, // B
               const float* __restrict__ hprev,          // B x H
               float* __restrict__ c_ws,                 // B x H (in/out)
               const unsigned short* __restrict__ WcT,   // [4096][2048] bf16
               const float* __restrict__ bias,           // 4H
               float* __restrict__ hout)                 // B x H (out[t])
{
    // LDS tile: 32 rows x 256 K of bf16, row stride 264 (528B: 16B pad -> 4-bank rotation)
    __shared__ __align__(16) unsigned short smem[32 * 264];

    const int tid  = threadIdx.x;
    const int lane = tid & 31;
    const int wave = tid >> 5;
    const int wr   = wave >> 2;   // 0..1  : 16-row group
    const int wc   = wave & 3;    // 0..3  : 16-col group
    const int m0   = blockIdx.x * 32;
    const int n0   = blockIdx.y * 64 + wc * 16;

    v8f acc[4];
#pragma unroll
    for (int g = 0; g < 4; ++g)
#pragma unroll
        for (int v = 0; v < 8; ++v) acc[g][v] = 0.0f;

    const int l16  = lane & 15;
    const int lhi  = lane >> 4;      // 0 or 1

    // ---- K loop: 8 chunks of 256 (chunks 0..3 = x, 4..7 = h) ----
    for (int kc = 0; kc < 8; ++kc) {
        __syncthreads();
        {
            const bool isH  = (kc >= 4);
            const int  koff = (kc & 3) * 256;
#pragma unroll
            for (int i = 0; i < 8; ++i) {
                int idx  = tid + i * 256;      // 0..2047 float4 slots (32 rows x 64)
                int r    = idx >> 6;           // 0..31
                int c4   = idx & 63;           // float4 within row
                int grow = m0 + r;
                float4 v;
                if (!isH) {
                    v = *(const float4*)(obs_t + (size_t)grow * DD + koff + c4 * 4);
                } else {
                    v = *(const float4*)(hprev + (size_t)grow * HH + koff + c4 * 4);
                    if (resets_t[grow]) v = make_float4(0.f, 0.f, 0.f, 0.f);
                }
                unsigned lo = (unsigned)f2bf(v.x) | ((unsigned)f2bf(v.y) << 16);
                unsigned hi = (unsigned)f2bf(v.z) | ((unsigned)f2bf(v.w) << 16);
                *(uint2*)&smem[r * 264 + c4 * 4] = make_uint2(lo, hi);
            }
        }
        __syncthreads();

        const int kg_base = kc * 256;
#pragma unroll
        for (int ks = 0; ks < 8; ++ks) {
            const int k0 = ks * 32;
            // A fragment (16x32 bf16): lane = row l16, 16 consecutive K per half-wave
            const int arow = wr * 16 + l16;
            const int kb   = lhi * 16;
            const unsigned short* ap = &smem[arow * 264 + k0 + kb];
            uint4 a_lo = *(const uint4*)(ap);
            uint4 a_hi = *(const uint4*)(ap + 8);
            v16bf afrag = make_frag(a_lo, a_hi);

            // B fragments (32x16 bf16): lane = col l16, 16 consecutive K per half-wave
            const size_t kgl = (size_t)(kg_base + k0 + lhi * 16);
#pragma unroll
            for (int g = 0; g < 4; ++g) {
                const size_t ncol = (size_t)(g * HH + n0 + l16);
                const unsigned short* bp = WcT + ncol * (size_t)KTOT + kgl;
                uint4 b_lo = *(const uint4*)(bp);
                uint4 b_hi = *(const uint4*)(bp + 8);
                v16bf bfrag = make_frag(b_lo, b_hi);
                acc[g] = __builtin_amdgcn_wmma_f32_16x16x32_bf16(
                    /*neg_a=*/false, afrag, /*neg_b=*/false, bfrag,
                    /*c_mod=*/(short)0, acc[g],
                    /*reuse_a=*/false, /*reuse_b=*/false);
            }
        }
    }

    // ---- fused gate nonlinearity + state update ----
    const int col   = n0 + l16;
    const int rbase = m0 + wr * 16 + (lhi << 3);
    const float bi = bias[col];
    const float bf = bias[HH + col];
    const float bg = bias[2 * HH + col];
    const float bo = bias[3 * HH + col];
#pragma unroll
    for (int v = 0; v < 8; ++v) {
        const int r = rbase + v;                    // C/D layout: M = v + 8*(lane/16)
        const size_t off = (size_t)r * HH + col;
        float gi = acc[0][v] + bi;
        float gf = acc[1][v] + bf;
        float gg = acc[2][v] + bg;
        float go = acc[3][v] + bo;
        float cp = c_ws[off];
        if (resets_t[r]) cp = 0.0f;
        float nc = sigmoidf_(gf) * cp + sigmoidf_(gi) * tanhf_(gg);
        float nh = sigmoidf_(go) * tanhf_(nc);
        c_ws[off] = nc;
        hout[off] = nh;
    }
}

// ---------------------------------------------------------------------------
extern "C" void kernel_launch(void* const* d_in, const int* in_sizes, int n_in,
                              void* d_out, int out_size, void* d_ws, size_t ws_size,
                              hipStream_t stream) {
    const float*         obs    = (const float*)d_in[0];          // T*B*D
    const unsigned char* resets = (const unsigned char*)d_in[1];  // T*B (bool)
    const float*         h0     = (const float*)d_in[2];          // B*H
    const float*         c0     = (const float*)d_in[3];          // B*H
    const float*         Wx     = (const float*)d_in[4];          // D*4H
    const float*         Wh     = (const float*)d_in[5];          // H*4H
    const float*         bias   = (const float*)d_in[6];          // 4H
    (void)in_sizes; (void)n_in; (void)out_size; (void)ws_size;

    float* out = (float*)d_out;                       // T*B*H
    float* hT  = out + (size_t)TT * BB * HH;          // B*H
    float* cT  = hT + (size_t)BB * HH;                // B*H

    // workspace layout: [ WcT bf16 16MB | c_ws f32 512KB ]
    unsigned short* WcT = (unsigned short*)d_ws;
    float* c_ws = (float*)((char*)d_ws + (size_t)NG * KTOT * sizeof(unsigned short));

    // one-time weight transpose/convert + c init
    lstm_prep_weights<<<(NG * KTOT) / 256, 256, 0, stream>>>(Wx, Wh, WcT);
    hipMemcpyAsync(c_ws, c0, (size_t)BB * HH * sizeof(float),
                   hipMemcpyDeviceToDevice, stream);

    // sequential scan: 512 launches, serialized on the stream
    for (int t = 0; t < TT; ++t) {
        const float* hprev = (t == 0) ? h0 : out + (size_t)(t - 1) * BB * HH;
        lstm_step<<<dim3(4, 16), 256, 0, stream>>>(
            obs + (size_t)t * BB * DD,
            resets + (size_t)t * BB,
            hprev, c_ws, WcT, bias,
            out + (size_t)t * BB * HH);
    }

    // final h/c
    hipMemcpyAsync(hT, out + (size_t)(TT - 1) * BB * HH,
                   (size_t)BB * HH * sizeof(float), hipMemcpyDeviceToDevice, stream);
    hipMemcpyAsync(cT, c_ws,
                   (size_t)BB * HH * sizeof(float), hipMemcpyDeviceToDevice, stream);
}
